// TextLSTM_34119220199992
// MI455X (gfx1250) — compile-verified, matrix operand order checked by reading, not run on hardware
//
#include <hip/hip_runtime.h>
#include <stdint.h>

// ---------------------------------------------------------------------------
// Types for CDNA5 WMMA (wave32, 16x16x32 bf16 -> f32)
// ---------------------------------------------------------------------------
typedef __attribute__((ext_vector_type(16))) __bf16 v16bf;
typedef __attribute__((ext_vector_type(8)))  __bf16 v8bf;
typedef __attribute__((ext_vector_type(8)))  float  v8f;

union ABfrag { v16bf v; v8bf h[2]; };

__device__ __forceinline__ unsigned short f32_to_bf16(float f) {
    union { float f; unsigned u; } x; x.f = f;
    unsigned u = x.u;
    u += 0x7FFFu + ((u >> 16) & 1u);   // round-to-nearest-even
    return (unsigned short)(u >> 16);
}

#define Hdim 1024
#define Bdim 64
#define Tdim 256
#define Cdim 32000

// ---------------------------------------------------------------------------
// fp32 -> bf16 convert with destination row stride (used to pack [Wi|Wh])
// ---------------------------------------------------------------------------
__global__ void cvt_pack_bf16(const float* __restrict__ src,
                              unsigned short* __restrict__ dst,
                              int rowlen, int dstStride, int total) {
    int idx = blockIdx.x * blockDim.x + threadIdx.x;
    if (idx >= total) return;
    int r = idx / rowlen, col = idx - r * rowlen;
    dst[(size_t)r * dstStride + col] = f32_to_bf16(src[idx]);
}

// ---------------------------------------------------------------------------
// Embedding gather: Xe[t][b][e] = bf16(emb[X[b][t]][e])
// ---------------------------------------------------------------------------
__global__ void embed_kernel(const int* __restrict__ X, const float* __restrict__ emb,
                             unsigned short* __restrict__ Xe) {
    int tb = blockIdx.x;               // t*64 + b
    int t = tb >> 6, b = tb & 63;
    int tok = X[(size_t)b * Tdim + t];
    const float* src = emb + (size_t)tok * 1024;
    unsigned short* dst = Xe + (size_t)tb * 1024;
    int e = threadIdx.x * 4;
    #pragma unroll
    for (int i = 0; i < 4; ++i) dst[e + i] = f32_to_bf16(src[e + i]);
}

// ---------------------------------------------------------------------------
// bias sums + zero-init h (buffer 0) and c
// ---------------------------------------------------------------------------
__global__ void prep_small(const float* __restrict__ bi1, const float* __restrict__ bh1,
                           const float* __restrict__ bi2, const float* __restrict__ bh2,
                           float* __restrict__ bsum, unsigned short* __restrict__ h0,
                           float* __restrict__ c) {
    int idx = blockIdx.x * blockDim.x + threadIdx.x;      // 131072 threads
    if (idx < 8192) {
        const float* bi = (idx < 4096) ? bi1 : bi2;
        const float* bh = (idx < 4096) ? bh1 : bh2;
        bsum[idx] = bi[idx & 4095] + bh[idx & 4095];
    }
    h0[idx] = 0;
    c[idx]  = 0.0f;
}

// ---------------------------------------------------------------------------
// One LSTM timestep, both directions (blockIdx.y), fused GEMM + cell update.
// Grid: (H/16, 2), 256 threads (8 waves). Each WG: 64 gate-rows x 64 batch,
// K = 2048 ([x_t | h_{t-1}] against packed [Wi | Wh]).
// ---------------------------------------------------------------------------
__global__ __launch_bounds__(256) void lstm_step_kernel(
    const unsigned short* __restrict__ Wcat,   // [2][4096][2048] bf16
    const unsigned short* __restrict__ Xe,     // [T][B][E] bf16
    const float* __restrict__ bsum,            // [2][4096]
    const unsigned short* __restrict__ h_in,   // [2][B][H] bf16
    unsigned short* __restrict__ h_out,        // [2][B][H] bf16
    float* __restrict__ c,                     // [2][B][H]
    float* __restrict__ o_last,                // [2][B][H]
    int t, int is_last)
{
    const int dir  = blockIdx.y;
    const int h0   = blockIdx.x * 16;
    const int xt   = (dir == 0) ? t : (Tdim - 1 - t);
    const int tid  = threadIdx.x;
    const int lane = tid & 31;
    const int wave = tid >> 5;
    const int l16  = lane & 15;
    const int khalf = (lane >> 4) * 8;     // per-lane K sub-block (ISA A layout)

    const int mi  = wave >> 1;             // gate tile 0..3 (i,f,g,o)
    const int ni0 = (wave & 1) * 2;        // batch tile base (0 or 2)

    __shared__ float gacc[64][64];         // [gate-unit][batch]

    // A rows: Wcat[dir][mi*H + h0 + l16][*]
    const unsigned short* arow =
        Wcat + ((size_t)dir * 4096 + (size_t)mi * Hdim + h0 + l16) * 2048 + khalf;

    // B rows (input columns): batch b = (ni0+j)*16 + l16
    const unsigned short* bx[2];
    const unsigned short* bh[2];
    #pragma unroll
    for (int j = 0; j < 2; ++j) {
        int b = (ni0 + j) * 16 + l16;
        bx[j] = Xe   + ((size_t)xt  * Bdim + b) * 1024 + khalf;
        bh[j] = h_in + ((size_t)dir * Bdim + b) * 1024 + khalf;
    }

    v8f acc0 = {}; v8f acc1 = {};

    #pragma unroll
    for (int seg = 0; seg < 2; ++seg) {
        const unsigned short* aseg = arow + seg * 1024;
        const unsigned short* b0   = seg ? bh[0] : bx[0];
        const unsigned short* b1   = seg ? bh[1] : bx[1];
        #pragma unroll 4
        for (int k = 0; k < 1024; k += 32) {
            ABfrag a, f0, f1;
            a.h[0]  = *(const v8bf*)(aseg + k);
            a.h[1]  = *(const v8bf*)(aseg + k + 16);
            f0.h[0] = *(const v8bf*)(b0 + k);
            f0.h[1] = *(const v8bf*)(b0 + k + 16);
            f1.h[0] = *(const v8bf*)(b1 + k);
            f1.h[1] = *(const v8bf*)(b1 + k + 16);
            acc0 = __builtin_amdgcn_wmma_f32_16x16x32_bf16(false, a.v, false, f0.v,
                                                           (short)0, acc0, false, false);
            acc1 = __builtin_amdgcn_wmma_f32_16x16x32_bf16(false, a.v, false, f1.v,
                                                           (short)0, acc1, false, false);
        }
    }

    // Spill accumulators: C/D layout -> m = mi*16 + (lane>=16)*8 + r, n = ni*16 + l16
    {
        int mrow = mi * 16 + (lane >> 4) * 8;
        #pragma unroll
        for (int r = 0; r < 8; ++r) {
            gacc[mrow + r][ni0 * 16 + l16]       = acc0[r];
            gacc[mrow + r][(ni0 + 1) * 16 + l16] = acc1[r];
        }
    }
    __syncthreads();

    // Fused LSTM cell update for this H-chunk (16 units x 64 batch).
    for (int item = tid; item < 1024; item += 256) {
        int b  = item & 63;
        int hh = item >> 6;
        const float* bs = bsum + (size_t)dir * 4096 + h0 + hh;
        float gi = gacc[ 0 + hh][b] + bs[0 * Hdim];
        float gf = gacc[16 + hh][b] + bs[1 * Hdim];
        float gg = gacc[32 + hh][b] + bs[2 * Hdim];
        float go = gacc[48 + hh][b] + bs[3 * Hdim];
        float i_t = 1.0f / (1.0f + __expf(-gi));
        float f_t = 1.0f / (1.0f + __expf(-gf));
        float g_t = tanhf(gg);
        float o_t = 1.0f / (1.0f + __expf(-go));
        size_t idx = ((size_t)dir * Bdim + b) * Hdim + h0 + hh;
        float cn = f_t * c[idx] + i_t * g_t;
        c[idx] = cn;
        h_out[idx] = f32_to_bf16(o_t * tanhf(cn));
        if (is_last) o_last[idx] = o_t;
    }
}

// ---------------------------------------------------------------------------
// ob[b][h] = bf16(o_fwd + o_bwd)
// ---------------------------------------------------------------------------
__global__ void sum_o_kernel(const float* __restrict__ o_last,
                             unsigned short* __restrict__ ob) {
    int idx = blockIdx.x * blockDim.x + threadIdx.x;      // 64*1024
    ob[idx] = f32_to_bf16(o_last[idx] + o_last[Bdim * Hdim + idx]);
}

// ---------------------------------------------------------------------------
// Classifier: logits[64][32000] = ob @ W^T + b  (bf16 WMMA, fp32 out)
// Grid: 500 WGs, each 64 classes x 64 batch, K = 1024.
// ---------------------------------------------------------------------------
__global__ __launch_bounds__(256) void cls_kernel(
    const unsigned short* __restrict__ Wbf,   // [32000][1024] bf16
    const unsigned short* __restrict__ ob,    // [64][1024] bf16
    const float* __restrict__ bias,           // [32000]
    float* __restrict__ out)                  // [64][32000]
{
    const int c0   = blockIdx.x * 64;
    const int tid  = threadIdx.x;
    const int lane = tid & 31;
    const int wave = tid >> 5;
    const int l16  = lane & 15;
    const int khalf = (lane >> 4) * 8;
    const int mi  = wave >> 1;
    const int ni0 = (wave & 1) * 2;

    const unsigned short* arow  = Wbf + (size_t)(c0 + mi * 16 + l16) * 1024 + khalf;
    const unsigned short* brow0 = ob + (size_t)(ni0 * 16 + l16) * 1024 + khalf;
    const unsigned short* brow1 = ob + (size_t)((ni0 + 1) * 16 + l16) * 1024 + khalf;

    v8f acc0 = {}; v8f acc1 = {};
    #pragma unroll 4
    for (int k = 0; k < 1024; k += 32) {
        ABfrag a, f0, f1;
        a.h[0]  = *(const v8bf*)(arow + k);
        a.h[1]  = *(const v8bf*)(arow + k + 16);
        f0.h[0] = *(const v8bf*)(brow0 + k);
        f0.h[1] = *(const v8bf*)(brow0 + k + 16);
        f1.h[0] = *(const v8bf*)(brow1 + k);
        f1.h[1] = *(const v8bf*)(brow1 + k + 16);
        acc0 = __builtin_amdgcn_wmma_f32_16x16x32_bf16(false, a.v, false, f0.v,
                                                       (short)0, acc0, false, false);
        acc1 = __builtin_amdgcn_wmma_f32_16x16x32_bf16(false, a.v, false, f1.v,
                                                       (short)0, acc1, false, false);
    }

    int clsBase = c0 + mi * 16 + (lane >> 4) * 8;
    int b0i = ni0 * 16 + l16;
    int b1i = (ni0 + 1) * 16 + l16;
    #pragma unroll
    for (int r = 0; r < 8; ++r) {
        float bb = bias[clsBase + r];
        out[(size_t)b0i * Cdim + clsBase + r] = acc0[r] + bb;
        out[(size_t)b1i * Cdim + clsBase + r] = acc1[r] + bb;
    }
}

// ---------------------------------------------------------------------------
// Host orchestration (graph-capture safe: only kernel launches on `stream`)
// ---------------------------------------------------------------------------
extern "C" void kernel_launch(void* const* d_in, const int* in_sizes, int n_in,
                              void* d_out, int out_size, void* d_ws, size_t ws_size,
                              hipStream_t stream) {
    (void)in_sizes; (void)n_in; (void)out_size; (void)ws_size;
    const int*   X    = (const int*)d_in[0];
    const float* emb  = (const float*)d_in[1];
    const float* Wi1  = (const float*)d_in[2];
    const float* Wh1  = (const float*)d_in[3];
    const float* bi1  = (const float*)d_in[4];
    const float* bh1  = (const float*)d_in[5];
    const float* Wi2  = (const float*)d_in[6];
    const float* Wh2  = (const float*)d_in[7];
    const float* bi2  = (const float*)d_in[8];
    const float* bh2  = (const float*)d_in[9];
    const float* W    = (const float*)d_in[10];
    const float* bcls = (const float*)d_in[11];

    char* ws = (char*)d_ws;
    size_t off = 0;
    auto carve = [&](size_t bytes) -> char* {
        char* p = ws + off;
        off = (off + bytes + 255) & ~(size_t)255;
        return p;
    };
    unsigned short* Wcat = (unsigned short*)carve((size_t)2 * 4096 * 2048 * 2); // 32 MB
    unsigned short* Wbf  = (unsigned short*)carve((size_t)Cdim * 1024 * 2);     // 62.5 MB
    unsigned short* Xe   = (unsigned short*)carve((size_t)Tdim * Bdim * 1024 * 2);
    unsigned short* hbuf = (unsigned short*)carve((size_t)2 * 2 * Bdim * Hdim * 2);
    float* cbuf = (float*)carve((size_t)2 * Bdim * Hdim * 4);
    float* obuf = (float*)carve((size_t)2 * Bdim * Hdim * 4);
    float* bsum = (float*)carve((size_t)2 * 4096 * 4);
    unsigned short* ob = (unsigned short*)carve((size_t)Bdim * Hdim * 2);

    // Pack [Wi | Wh] -> Wcat (bf16), convert classifier W -> bf16
    int nW = 4096 * 1024;
    cvt_pack_bf16<<<(nW + 255) / 256, 256, 0, stream>>>(Wi1, Wcat, 1024, 2048, nW);
    cvt_pack_bf16<<<(nW + 255) / 256, 256, 0, stream>>>(Wh1, Wcat + 1024, 1024, 2048, nW);
    cvt_pack_bf16<<<(nW + 255) / 256, 256, 0, stream>>>(Wi2, Wcat + (size_t)4096 * 2048, 1024, 2048, nW);
    cvt_pack_bf16<<<(nW + 255) / 256, 256, 0, stream>>>(Wh2, Wcat + (size_t)4096 * 2048 + 1024, 1024, 2048, nW);
    int nC = Cdim * 1024;
    cvt_pack_bf16<<<(nC + 255) / 256, 256, 0, stream>>>(W, Wbf, 1024, 1024, nC);

    embed_kernel<<<Tdim * Bdim, 256, 0, stream>>>(X, emb, Xe);
    prep_small<<<512, 256, 0, stream>>>(bi1, bh1, bi2, bh2, bsum, hbuf, cbuf);

    // Recurrence: 256 serial steps, h double-buffered across launches.
    const size_t hstride = (size_t)2 * Bdim * Hdim;
    for (int t = 0; t < Tdim; ++t) {
        unsigned short* h_in  = hbuf + (size_t)(t & 1) * hstride;
        unsigned short* h_out = hbuf + (size_t)((t + 1) & 1) * hstride;
        lstm_step_kernel<<<dim3(Hdim / 16, 2), 256, 0, stream>>>(
            Wcat, Xe, bsum, h_in, h_out, cbuf, obuf, t, (t == Tdim - 1) ? 1 : 0);
    }

    sum_o_kernel<<<(Bdim * Hdim) / 256, 256, 0, stream>>>(obuf, ob);
    cls_kernel<<<Cdim / 64, 256, 0, stream>>>(Wbf, ob, bcls, (float*)d_out);
}
